// TSNet_9912784520003
// MI455X (gfx1250) — matrix operand order repaired and checked by reading, not compiled, not run on hardware
//
#include <hip/hip_runtime.h>
#include <hip/hip_bf16.h>

#define DGRID 64

typedef _Float16 half8 __attribute__((ext_vector_type(8)));   // 16B
typedef _Float16 v16h  __attribute__((ext_vector_type(16)));  // A/B fragment
typedef float    v8f   __attribute__((ext_vector_type(8)));   // C/D fragment

// ---------------- rulebook construction ----------------

__global__ void init_grid_k(int* __restrict__ grid, int total) {
    int i = blockIdx.x * blockDim.x + threadIdx.x;
    if (i < total) grid[i] = -1;
}

__global__ void scatter_k(const int* __restrict__ coords, int* __restrict__ grid, int n) {
    int i = blockIdx.x * blockDim.x + threadIdx.x;
    if (i >= n) return;
    int x = coords[3 * i], y = coords[3 * i + 1], z = coords[3 * i + 2];
    grid[(x * DGRID + y) * DGRID + z] = i;   // coords are unique -> deterministic
}

__global__ void rulebook_k(const int* __restrict__ coords, const int* __restrict__ grid,
                           int* __restrict__ nbr, int n) {
    int i = blockIdx.x * blockDim.x + threadIdx.x;
    if (i >= n) return;
    int x = coords[3 * i], y = coords[3 * i + 1], z = coords[3 * i + 2];
#pragma unroll
    for (int k = 0; k < 27; ++k) {
        // matches jnp.meshgrid(r,r,r,indexing='ij'): dx major, dz minor
        int dx = k / 9 - 1, dy = (k / 3) % 3 - 1, dz = k % 3 - 1;
        int nx = x + dx, ny = y + dy, nz = z + dz;
        int v = -1;
        if (nx >= 0 && nx < DGRID && ny >= 0 && ny < DGRID && nz >= 0 && nz < DGRID)
            v = grid[(nx * DGRID + ny) * DGRID + nz];
        nbr[k * n + i] = v;
    }
}

// ---------------- precision conversion / scratch init ----------------

__global__ void zero_row_k(_Float16* __restrict__ z) {
    z[threadIdx.x] = (_Float16)0.0f;
}

// features [N,3] f32 -> [N,32] f16 zero-padded
__global__ void cvt_feat_k(const float* __restrict__ f, _Float16* __restrict__ x, int n) {
    int i = blockIdx.x * blockDim.x + threadIdx.x;
    if (i >= n * 32) return;
    int p = i >> 5, c = i & 31;
    x[i] = (c < 3) ? (_Float16)f[p * 3 + c] : (_Float16)0.0f;
}

// W [27,cin,cout] f32 -> Wt [27,cout,cinPad] f16 (transposed: B fragments contiguous)
__global__ void cvt_w_k(const float* __restrict__ w, _Float16* __restrict__ wt,
                        int cin, int cinPad, int cout) {
    int i = blockIdx.x * blockDim.x + threadIdx.x;
    int total = 27 * cout * cinPad;
    if (i >= total) return;
    int ci = i % cinPad;
    int t  = i / cinPad;
    int co = t % cout;
    int k  = t / cout;
    float v = (ci < cin) ? w[(k * cin + ci) * cout + co] : 0.0f;
    wt[i] = (_Float16)v;
}

// ---------------- WMMA gather-GEMM layer ----------------

static __device__ inline half8 ldh8(const _Float16* p) { return *(const half8*)p; }

static __device__ inline v16h pack16(half8 lo, half8 hi) {
    v16h r;
#pragma unroll
    for (int e = 0; e < 8; ++e) { r[e] = lo[e]; r[8 + e] = hi[e]; }
    return r;
}

static __device__ inline v8f wmma_f16(v16h a, v16h b, v8f c) {
    return __builtin_amdgcn_wmma_f32_16x16x32_f16(false, a, false, b, (short)0, c,
                                                  false, false);
}

struct Frag4 { v16h A, B0, B1, B2, B3; };
struct Frag2 { v16h A, B0, B1; };

// One wave per block: 16 points x 64 output channels, four accumulators.
// Ping-pong software pipeline: fragments alternate between two register sets
// (no loop-carried copies); each WMMA group consumes fragments whose loads
// were issued one iteration earlier, so the single wave overlaps L2 gather
// latency with the matrix pipe. Running pointers (+32 halves/chunk) keep the
// per-iteration address math to five v_add_nc_u64 with immediate load offsets.
__global__ __launch_bounds__(32) void subm_gemm4_k(
    const _Float16* __restrict__ xin,    // [n, cinPad] f16 activations
    const _Float16* __restrict__ wt,     // [27, cout, cinPad] f16 transposed weights
    const int*      __restrict__ nbr,    // [27, n] neighbor index or -1
    const _Float16* __restrict__ zeroRow,// cinPad zeros: null-neighbor target
    float*          __restrict__ outF32,
    _Float16*       __restrict__ outF16,
    int n, int cinPad, int cout, int writeF32)
{
    const int lane    = threadIdx.x & 31;
    const int r       = lane & 15;
    const int hi      = lane >> 4;
    const int ptile   = blockIdx.x;
    const int p       = ptile * 16 + r;        // A-matrix row (point)
    const int colBase = blockIdx.y * 64;
    const int col     = colBase + r;           // B column (tile 0)

    v8f acc0 = {}, acc1 = {}, acc2 = {}, acc3 = {};

    const int nChunks = cinPad >> 5;
    const int total   = 27 * nChunks;

    // pipeline state: (k, c) with neighbor index prefetched one k ahead
    int k   = 0, c = 0;
    int nb  = (p < n) ? nbr[p] : -1;
    int nbN = (p < n) ? nbr[n + p] : -1;
    // lane-constant offsets folded into running pointers:
    //   A: xp = row + hi*16  -> loads at xp, xp+8    (contiguous k = 16*hi + s, same as B)
    //   B: wp = col + hi*16  -> loads at wp, wp+8    (contiguous k = 16*hi + s, same as A)
    const _Float16* xp  = ((nb >= 0) ? (xin + (size_t)nb * cinPad) : zeroRow) + hi * 16;
    const _Float16* wp0 = wt + ((size_t)0 * cout + col) * cinPad + hi * 16;
    const _Float16* wp1 = wp0 + 16 * cinPad;
    const _Float16* wp2 = wp0 + 32 * cinPad;
    const _Float16* wp3 = wp0 + 48 * cinPad;

    auto advance = [&]() {
        int cn = c + 1;
        if (cn == nChunks) {               // uniform branch: k transition
            cn = 0;
            ++k;
            xp = ((nbN >= 0) ? (xin + (size_t)nbN * cinPad) : zeroRow) + hi * 16;
            int kpre = (k + 1 < 27) ? (k + 1) : 26;   // prefetch next k's index
            nbN = (p < n) ? nbr[kpre * n + p] : -1;
            const _Float16* w0 = wt + ((size_t)k * cout + col) * cinPad + hi * 16;
            wp0 = w0;
            wp1 = w0 + 16 * cinPad;
            wp2 = w0 + 32 * cinPad;
            wp3 = w0 + 48 * cinPad;
        } else {
            xp += 32; wp0 += 32; wp1 += 32; wp2 += 32; wp3 += 32;
        }
        c = cn;
    };
    auto loadFrag = [&](Frag4& f) {
        f.A  = pack16(ldh8(xp), ldh8(xp + 8));
        f.B0 = pack16(ldh8(wp0), ldh8(wp0 + 8));
        f.B1 = pack16(ldh8(wp1), ldh8(wp1 + 8));
        f.B2 = pack16(ldh8(wp2), ldh8(wp2 + 8));
        f.B3 = pack16(ldh8(wp3), ldh8(wp3 + 8));
    };
    auto domma = [&](const Frag4& f) {
        acc0 = wmma_f16(f.A, f.B0, acc0);
        acc1 = wmma_f16(f.A, f.B1, acc1);
        acc2 = wmma_f16(f.A, f.B2, acc2);
        acc3 = wmma_f16(f.A, f.B3, acc3);
    };

    Frag4 f0, f1;
    loadFrag(f0);                                  // iteration 0
    for (int t = 0; t + 1 < total; t += 2) {
        advance();                                 // -> t+1
        loadFrag(f1);
        domma(f0);                                 // t
        if (t + 2 < total) {
            advance();                             // -> t+2
            loadFrag(f0);
        }
        domma(f1);                                 // t+1
    }
    if (total & 1) domma(f0);                      // odd tail

    // C/D layout: VGPR v -> row v (lanes 0-15) / v+8 (lanes 16-31), col = lane&15
    const int c0 = colBase + r;
#pragma unroll
    for (int v = 0; v < 8; ++v) {
        int row = ptile * 16 + v + hi * 8;
        if (row < n) {
            if (writeF32) {
                float* orow = outF32 + (size_t)row * cout;
                orow[c0]      = acc0[v];
                orow[c0 + 16] = acc1[v];
                orow[c0 + 32] = acc2[v];
                orow[c0 + 48] = acc3[v];
            } else {
                _Float16* orow = outF16 + (size_t)row * cout;
                orow[c0]      = (_Float16)acc0[v];
                orow[c0 + 16] = (_Float16)acc1[v];
                orow[c0 + 32] = (_Float16)acc2[v];
                orow[c0 + 48] = (_Float16)acc3[v];
            }
        }
    }
}

// Overhang kernel: last 32 output columns for layers with cout % 64 == 32.
__global__ __launch_bounds__(32) void subm_gemm2_k(
    const _Float16* __restrict__ xin,
    const _Float16* __restrict__ wt,
    const int*      __restrict__ nbr,
    const _Float16* __restrict__ zeroRow,
    float*          __restrict__ outF32,
    _Float16*       __restrict__ outF16,
    int n, int cinPad, int cout, int colBase, int writeF32)
{
    const int lane  = threadIdx.x & 31;
    const int r     = lane & 15;
    const int hi    = lane >> 4;
    const int ptile = blockIdx.x;
    const int p     = ptile * 16 + r;
    const int col   = colBase + r;

    v8f acc0 = {}, acc1 = {};

    const int nChunks = cinPad >> 5;
    const int total   = 27 * nChunks;

    int k   = 0, c = 0;
    int nb  = (p < n) ? nbr[p] : -1;
    int nbN = (p < n) ? nbr[n + p] : -1;
    const _Float16* xp  = ((nb >= 0) ? (xin + (size_t)nb * cinPad) : zeroRow) + hi * 16;
    const _Float16* wp0 = wt + ((size_t)0 * cout + col) * cinPad + hi * 16;
    const _Float16* wp1 = wp0 + 16 * cinPad;

    auto advance = [&]() {
        int cn = c + 1;
        if (cn == nChunks) {
            cn = 0;
            ++k;
            xp = ((nbN >= 0) ? (xin + (size_t)nbN * cinPad) : zeroRow) + hi * 16;
            int kpre = (k + 1 < 27) ? (k + 1) : 26;
            nbN = (p < n) ? nbr[kpre * n + p] : -1;
            const _Float16* w0 = wt + ((size_t)k * cout + col) * cinPad + hi * 16;
            wp0 = w0;
            wp1 = w0 + 16 * cinPad;
        } else {
            xp += 32; wp0 += 32; wp1 += 32;
        }
        c = cn;
    };
    auto loadFrag = [&](Frag2& f) {
        f.A  = pack16(ldh8(xp), ldh8(xp + 8));
        f.B0 = pack16(ldh8(wp0), ldh8(wp0 + 8));
        f.B1 = pack16(ldh8(wp1), ldh8(wp1 + 8));
    };
    auto domma = [&](const Frag2& f) {
        acc0 = wmma_f16(f.A, f.B0, acc0);
        acc1 = wmma_f16(f.A, f.B1, acc1);
    };

    Frag2 f0, f1;
    loadFrag(f0);
    for (int t = 0; t + 1 < total; t += 2) {
        advance();
        loadFrag(f1);
        domma(f0);
        if (t + 2 < total) {
            advance();
            loadFrag(f0);
        }
        domma(f1);
    }
    if (total & 1) domma(f0);

    const int c0 = colBase + r;
#pragma unroll
    for (int v = 0; v < 8; ++v) {
        int row = ptile * 16 + v + hi * 8;
        if (row < n) {
            if (writeF32) {
                float* orow = outF32 + (size_t)row * cout;
                orow[c0]      = acc0[v];
                orow[c0 + 16] = acc1[v];
            } else {
                _Float16* orow = outF16 + (size_t)row * cout;
                orow[c0]      = (_Float16)acc0[v];
                orow[c0 + 16] = (_Float16)acc1[v];
            }
        }
    }
}

// ---------------- host orchestration ----------------

extern "C" void kernel_launch(void* const* d_in, const int* in_sizes, int n_in,
                              void* d_out, int out_size, void* d_ws, size_t ws_size,
                              hipStream_t stream) {
    static const int CH[14] = {3, 64, 64, 96, 96, 128, 160, 160, 192, 192, 224, 224, 256, 256};

    const float* features = (const float*)d_in[0];
    const int*   coords   = (const int*)d_in[1];
    const int n = in_sizes[0] / 3;          // 20000

    char*  ws = (char*)d_ws;
    size_t o  = 0;
    auto carve = [&](size_t bytes) -> void* {
        void* p = ws + o;
        o += (bytes + 255) & ~(size_t)255;
        return p;
    };
    const int totalGrid = DGRID * DGRID * DGRID;
    int*      gridBuf = (int*)      carve((size_t)totalGrid * 4);
    int*      nbrBuf  = (int*)      carve((size_t)27 * n * 4);
    _Float16* xA      = (_Float16*) carve((size_t)n * 256 * 2);
    _Float16* xB      = (_Float16*) carve((size_t)n * 256 * 2);
    _Float16* wtBuf   = (_Float16*) carve((size_t)27 * 256 * 256 * 2);
    _Float16* zeroRow = (_Float16*) carve((size_t)256 * 2);

    // rulebook (built once, reused by all 13 layers)
    init_grid_k<<<(totalGrid + 255) / 256, 256, 0, stream>>>(gridBuf, totalGrid);
    scatter_k  <<<(n + 255) / 256, 256, 0, stream>>>(coords, gridBuf, n);
    rulebook_k <<<(n + 255) / 256, 256, 0, stream>>>(coords, gridBuf, nbrBuf, n);

    zero_row_k<<<1, 256, 0, stream>>>(zeroRow);
    cvt_feat_k<<<((n * 32) + 255) / 256, 256, 0, stream>>>(features, xA, n);

    _Float16* cur = xA;
    _Float16* nxt = xB;
    const int ptiles = (n + 15) / 16;
    for (int l = 0; l < 13; ++l) {
        const int cin    = CH[l];
        const int cout   = CH[l + 1];
        const int cinPad = (l == 0) ? 32 : cin;
        const float* w = (const float*)d_in[2 + l];

        const int totalW = 27 * cout * cinPad;
        cvt_w_k<<<(totalW + 255) / 256, 256, 0, stream>>>(w, wtBuf, cin, cinPad, cout);

        const int writeF32 = (l == 12);
        const int nFull = cout / 64;
        if (nFull > 0) {
            subm_gemm4_k<<<dim3(ptiles, nFull), 32, 0, stream>>>(
                cur, wtBuf, nbrBuf, zeroRow, (float*)d_out, nxt,
                n, cinPad, cout, writeF32);
        }
        if (cout % 64) {                       // 32-column overhang (96/160/224)
            subm_gemm2_k<<<dim3(ptiles, 1), 32, 0, stream>>>(
                cur, wtBuf, nbrBuf, zeroRow, (float*)d_out, nxt,
                n, cinPad, cout, cout - 32, writeF32);
        }
        _Float16* t = cur; cur = nxt; nxt = t;
    }
}
